// MambaBlock_29463475651160
// MI455X (gfx1250) — compile-verified, matrix operand order checked by reading, not run on hardware
//
#include <hip/hip_runtime.h>
#include <hip/hip_bf16.h>

typedef __attribute__((ext_vector_type(16))) _Float16 v16h;
typedef __attribute__((ext_vector_type(8)))  _Float16 v8h;
typedef __attribute__((ext_vector_type(4)))  _Float16 v4h;
typedef __attribute__((ext_vector_type(8)))  float    v8f;

#define D_MODEL  1024
#define D_STATE  16
#define D_CONV   4
#define D_INNER  2048
#define DT_RANK  64
#define SEQ_L    1024
#define XPROJ_N  96   // DT_RANK + 2*D_STATE

__device__ __forceinline__ float siluf(float x) { return x / (1.0f + __expf(-x)); }
__device__ __forceinline__ float softplusf(float x) {
  return (x > 20.0f) ? x : log1pf(__expf(x));
}

// ---------------------------------------------------------------------------
// WMMA fragment load. For C = A(MxK,row-major) * B(NxK,row-major)^T both
// operands use the CDNA5 16-bit A-matrix 16x32 layout:
//   lane 0-15 : row M=lane,    halves 0-7 = K[0..7],  halves 8-15 = K[16..23]
//   lane 16-31: row M=lane-16, halves 0-7 = K[8..15], halves 8-15 = K[24..31]
// (B^T columns == B rows, so the same addressing applies to B.)
// ---------------------------------------------------------------------------
__device__ __forceinline__ v16h load_frag(const _Float16* __restrict__ base,
                                          int ld, int row, int k0, int lane) {
  const int r  = row + (lane & 15);
  const int kk = k0 + ((lane >> 4) << 3);
  const _Float16* p = base + (long)r * ld + kk;
  union { v16h v; v8h h[2]; } t;
  t.h[0] = *(const v8h*)(p);
  t.h[1] = *(const v8h*)(p + 16);
  return t.v;
}

// ---------------------------------------------------------------------------
// 64x64 tile per wave: 4x4 grid of 16x16 WMMA tiles, 16 WMMAs per K=32 step.
// Arithmetic intensity ~32 FLOP/byte of fragment traffic (2x the 32x32 tile).
// Requires M % 64 == 0 && N % 64 == 0.
// ---------------------------------------------------------------------------
__global__ __launch_bounds__(32)
void gemm64_wmma_abT(const _Float16* __restrict__ A, const _Float16* __restrict__ B,
                     float* __restrict__ C, const float* __restrict__ bias,
                     int M, int N, int K) {
  const int lane = threadIdx.x;
  const int nb = blockIdx.x * 64;
  const int mb = blockIdx.y * 64;

  v8f acc[4][4];
#pragma unroll
  for (int i = 0; i < 4; ++i)
#pragma unroll
    for (int j = 0; j < 4; ++j) acc[i][j] = (v8f){};

  for (int k0 = 0; k0 < K; k0 += 32) {
    v16h a[4], b[4];
#pragma unroll
    for (int i = 0; i < 4; ++i) a[i] = load_frag(A, K, mb + 16 * i, k0, lane);
#pragma unroll
    for (int j = 0; j < 4; ++j) b[j] = load_frag(B, K, nb + 16 * j, k0, lane);
#pragma unroll
    for (int i = 0; i < 4; ++i)
#pragma unroll
      for (int j = 0; j < 4; ++j)
        acc[i][j] = __builtin_amdgcn_wmma_f32_16x16x32_f16(
            false, a[i], false, b[j], (short)0, acc[i][j], false, false);
  }

  // C/D layout: VGPR e, lanes 0-15 -> (M=e, N=lane); lanes 16-31 -> (M=e+8, N=lane-16)
  const int col = lane & 15;
  const int hi  = lane >> 4;
#pragma unroll
  for (int j = 0; j < 4; ++j) {
    const int c0 = nb + 16 * j + col;
    const float bv = bias ? bias[c0] : 0.0f;
#pragma unroll
    for (int i = 0; i < 4; ++i) {
      const int r0 = mb + 16 * i + 8 * hi;
#pragma unroll
      for (int e = 0; e < 8; ++e)
        C[(long)(r0 + e) * N + c0] = acc[i][j][e] + bv;
    }
  }
}

// 32x32 tile per wave (for N not divisible by 64, e.g. the 96-wide x_dbl GEMM).
__global__ __launch_bounds__(32)
void gemm_wmma_abT(const _Float16* __restrict__ A, const _Float16* __restrict__ B,
                   float* __restrict__ C, const float* __restrict__ bias,
                   int M, int N, int K) {
  const int lane = threadIdx.x;
  const int nb = blockIdx.x * 32;
  const int mb = blockIdx.y * 32;

  v8f acc00 = {}; v8f acc01 = {}; v8f acc10 = {}; v8f acc11 = {};
  for (int k0 = 0; k0 < K; k0 += 32) {
    v16h a0 = load_frag(A, K, mb,      k0, lane);
    v16h a1 = load_frag(A, K, mb + 16, k0, lane);
    v16h b0 = load_frag(B, K, nb,      k0, lane);
    v16h b1 = load_frag(B, K, nb + 16, k0, lane);
    acc00 = __builtin_amdgcn_wmma_f32_16x16x32_f16(false, a0, false, b0, (short)0, acc00, false, false);
    acc01 = __builtin_amdgcn_wmma_f32_16x16x32_f16(false, a0, false, b1, (short)0, acc01, false, false);
    acc10 = __builtin_amdgcn_wmma_f32_16x16x32_f16(false, a1, false, b0, (short)0, acc10, false, false);
    acc11 = __builtin_amdgcn_wmma_f32_16x16x32_f16(false, a1, false, b1, (short)0, acc11, false, false);
  }

  const int col = lane & 15;
  const int hi  = lane >> 4;
  const float bv0 = bias ? bias[nb + col]      : 0.0f;
  const float bv1 = bias ? bias[nb + 16 + col] : 0.0f;
#pragma unroll
  for (int j = 0; j < 8; ++j) {
    const int r0 = mb + j + 8 * hi;
    const int r1 = mb + 16 + j + 8 * hi;
    C[(long)r0 * N + nb + col]      = acc00[j] + bv0;
    C[(long)r0 * N + nb + 16 + col] = acc01[j] + bv1;
    C[(long)r1 * N + nb + col]      = acc10[j] + bv0;
    C[(long)r1 * N + nb + 16 + col] = acc11[j] + bv1;
  }
}

// ---------------------------------------------------------------------------
// Elementwise / helper kernels
// ---------------------------------------------------------------------------
// n must be a multiple of 4 (all tensor sizes here are).
__global__ void cvt_h_kernel(const float* __restrict__ s, _Float16* __restrict__ d, int n4) {
  int i = blockIdx.x * blockDim.x + threadIdx.x;
  if (i >= n4) return;
  float4 v = ((const float4*)s)[i];
  v4h o; o.x = (_Float16)v.x; o.y = (_Float16)v.y; o.z = (_Float16)v.z; o.w = (_Float16)v.w;
  ((v4h*)d)[i] = o;
}

__global__ void cvt_h_fliprows_kernel(const float* __restrict__ s, _Float16* __restrict__ d,
                                      int rows, int cols) {
  int i = blockIdx.x * blockDim.x + threadIdx.x;  // element-group of 4
  int n4 = rows * cols / 4;
  if (i >= n4) return;
  int c4 = cols / 4;
  int r = i / c4, c = i % c4;
  float4 v = ((const float4*)s)[(long)(rows - 1 - r) * c4 + c];
  v4h o; o.x = (_Float16)v.x; o.y = (_Float16)v.y; o.z = (_Float16)v.z; o.w = (_Float16)v.w;
  ((v4h*)d)[i] = o;
}

// Depthwise causal conv (k=4) + bias + SiLU. xi = xz[:, 0:D_INNER].
__global__ void conv_silu_kernel(const float* __restrict__ xz,
                                 const float* __restrict__ cw,
                                 const float* __restrict__ cb,
                                 float* __restrict__ u,
                                 _Float16* __restrict__ uh) {
  int idx = blockIdx.x * blockDim.x + threadIdx.x;
  if (idx >= SEQ_L * D_INNER) return;
  int l = idx / D_INNER, d = idx % D_INNER;
  float s = cb[d];
#pragma unroll
  for (int j = 0; j < D_CONV; ++j) {
    int t = l - (D_CONV - 1) + j;
    if (t >= 0) s += cw[d * D_CONV + j] * xz[(long)t * (2 * D_INNER) + d];
  }
  float v = siluf(s);
  u[idx]  = v;
  uh[idx] = (_Float16)v;
}

__global__ void slice_dt_kernel(const float* __restrict__ xdbl, _Float16* __restrict__ dth) {
  int i = blockIdx.x * blockDim.x + threadIdx.x;
  if (i >= SEQ_L * DT_RANK) return;
  int l = i / DT_RANK, r = i % DT_RANK;
  dth[i] = (_Float16)xdbl[l * XPROJ_N + r];
}

// Sequential selective scan: one thread per channel, 16 states in registers,
// B/C rows broadcast via LDS each timestep.
__global__ __launch_bounds__(256)
void scan_kernel(const float* __restrict__ dtlin, const float* __restrict__ dt_b,
                 const float* __restrict__ Alog, const float* __restrict__ u,
                 const float* __restrict__ xdbl, float* __restrict__ y) {
  __shared__ float sBC[2 * D_STATE];
  const int d = blockIdx.x * blockDim.x + threadIdx.x;

  float A[D_STATE], h[D_STATE];
#pragma unroll
  for (int n = 0; n < D_STATE; ++n) {
    A[n] = -__expf(Alog[d * D_STATE + n]);
    h[n] = 0.0f;
  }
  const float dtb = dt_b[d];

  for (int l = 0; l < SEQ_L; ++l) {
    if (threadIdx.x < 2 * D_STATE)
      sBC[threadIdx.x] = xdbl[l * XPROJ_N + DT_RANK + threadIdx.x];
    __syncthreads();

    const float dl = softplusf(dtlin[(long)l * D_INNER + d] + dtb);
    const float ul = u[(long)l * D_INNER + d];
    float acc = 0.0f;
#pragma unroll
    for (int n = 0; n < D_STATE; ++n) {
      const float a = __expf(dl * A[n]);
      h[n] = a * h[n] + dl * sBC[n] * ul;
      acc += h[n] * sBC[D_STATE + n];
    }
    y[(long)l * D_INNER + d] = acc;
    __syncthreads();
  }
}

// yz = (y + u*Dp) * silu(z), z = xz[:, D_INNER:2*D_INNER]; emit f16.
__global__ void gate_kernel(const float* __restrict__ y, const float* __restrict__ u,
                            const float* __restrict__ Dp, const float* __restrict__ xz,
                            _Float16* __restrict__ yzh) {
  int idx = blockIdx.x * blockDim.x + threadIdx.x;
  if (idx >= SEQ_L * D_INNER) return;
  int l = idx / D_INNER, d = idx % D_INNER;
  float z = xz[(long)l * (2 * D_INNER) + D_INNER + d];
  float v = (y[idx] + u[idx] * Dp[d]) * siluf(z);
  yzh[idx] = (_Float16)v;
}

// concat[l] = [mamba_fw[l] + x[l], mamba_bw_flipped[L-1-l] + x[l]]  (f16)
__global__ void concat_kernel(const float* __restrict__ mfw, const float* __restrict__ mbw,
                              const float* __restrict__ x, _Float16* __restrict__ ch) {
  int i = blockIdx.x * blockDim.x + threadIdx.x;
  if (i >= SEQ_L * D_MODEL) return;
  int l = i / D_MODEL, c = i % D_MODEL;
  float xv = x[(long)l * D_MODEL + c];
  ch[(long)l * (2 * D_MODEL) + c] =
      (_Float16)(mfw[(long)l * D_MODEL + c] + xv);
  ch[(long)l * (2 * D_MODEL) + D_MODEL + c] =
      (_Float16)(mbw[(long)(SEQ_L - 1 - l) * D_MODEL + c] + xv);
}

__global__ __launch_bounds__(256)
void layernorm_kernel(const float* __restrict__ p, const float* __restrict__ w,
                      const float* __restrict__ b, float* __restrict__ out) {
  __shared__ float red[256];
  const int r = blockIdx.x;
  const int tid = threadIdx.x;

  float s = 0.0f;
  for (int c = tid; c < D_MODEL; c += 256) s += p[(long)r * D_MODEL + c];
  red[tid] = s; __syncthreads();
  for (int k = 128; k > 0; k >>= 1) { if (tid < k) red[tid] += red[tid + k]; __syncthreads(); }
  const float mu = red[0] / (float)D_MODEL;
  __syncthreads();

  float v = 0.0f;
  for (int c = tid; c < D_MODEL; c += 256) {
    float t = p[(long)r * D_MODEL + c] - mu; v += t * t;
  }
  red[tid] = v; __syncthreads();
  for (int k = 128; k > 0; k >>= 1) { if (tid < k) red[tid] += red[tid + k]; __syncthreads(); }
  const float rstd = rsqrtf(red[0] / (float)D_MODEL + 1e-5f);

  for (int c = tid; c < D_MODEL; c += 256)
    out[(long)r * D_MODEL + c] = (p[(long)r * D_MODEL + c] - mu) * rstd * w[c] + b[c];
}

// ---------------------------------------------------------------------------
// Host-side orchestration
// ---------------------------------------------------------------------------
extern "C" void kernel_launch(void* const* d_in, const int* in_sizes, int n_in,
                              void* d_out, int out_size, void* d_ws, size_t ws_size,
                              hipStream_t stream) {
  (void)in_sizes; (void)n_in; (void)out_size; (void)ws_size;

  const float* x = (const float*)d_in[0];
  struct Params {
    const float *in_w, *conv_w, *conv_b, *xproj_w, *dt_w, *dt_b, *Alog, *Dp, *out_w;
  };
  auto getp = [&](int base) {
    Params p;
    p.in_w    = (const float*)d_in[base + 0];
    p.conv_w  = (const float*)d_in[base + 1];
    p.conv_b  = (const float*)d_in[base + 2];
    p.xproj_w = (const float*)d_in[base + 3];
    p.dt_w    = (const float*)d_in[base + 4];
    p.dt_b    = (const float*)d_in[base + 5];
    p.Alog    = (const float*)d_in[base + 6];
    p.Dp      = (const float*)d_in[base + 7];
    p.out_w   = (const float*)d_in[base + 8];
    return p;
  };
  const Params fw = getp(1);
  const Params bw = getp(10);
  const float* proj_w = (const float*)d_in[19];
  const float* proj_b = (const float*)d_in[20];
  const float* ln_w   = (const float*)d_in[21];
  const float* ln_b   = (const float*)d_in[22];

  // -------- workspace carving (256B aligned) --------
  char* ws = (char*)d_ws;
  size_t off = 0;
  auto carve = [&](size_t bytes) -> void* {
    void* p = ws + off;
    off = (off + bytes + 255) & ~(size_t)255;
    return p;
  };
  // per-direction scratch (reused between fw/bw passes)
  _Float16* xh     = (_Float16*)carve((size_t)SEQ_L * D_MODEL * 2);
  _Float16* inwh   = (_Float16*)carve((size_t)2 * D_INNER * D_MODEL * 2);
  _Float16* outwh  = (_Float16*)carve((size_t)D_MODEL * D_INNER * 2);
  _Float16* xprojh = (_Float16*)carve((size_t)XPROJ_N * D_INNER * 2);
  _Float16* dtwh   = (_Float16*)carve((size_t)D_INNER * DT_RANK * 2);
  float*    xz     = (float*)carve((size_t)SEQ_L * 2 * D_INNER * 4);
  float*    u      = (float*)carve((size_t)SEQ_L * D_INNER * 4);
  _Float16* uh     = (_Float16*)carve((size_t)SEQ_L * D_INNER * 2);
  float*    xdbl   = (float*)carve((size_t)SEQ_L * XPROJ_N * 4);
  _Float16* dth    = (_Float16*)carve((size_t)SEQ_L * DT_RANK * 2);
  float*    dtlin  = (float*)carve((size_t)SEQ_L * D_INNER * 4);
  float*    y      = (float*)carve((size_t)SEQ_L * D_INNER * 4);
  _Float16* yzh    = (_Float16*)carve((size_t)SEQ_L * D_INNER * 2);
  // persistent across directions
  float*    mout_fw = (float*)carve((size_t)SEQ_L * D_MODEL * 4);
  float*    mout_bw = (float*)carve((size_t)SEQ_L * D_MODEL * 4);
  _Float16* conch   = (_Float16*)carve((size_t)SEQ_L * 2 * D_MODEL * 2);
  float*    pout    = (float*)carve((size_t)SEQ_L * D_MODEL * 4);
  _Float16* projwh  = (_Float16*)carve((size_t)D_MODEL * 2 * D_MODEL * 2);

  const int TB = 256;
  auto blocks  = [&](int n) { return (n + TB - 1) / TB; };
  auto blocks4 = [&](int n) { return (n / 4 + TB - 1) / TB; };

  auto run_dir = [&](const Params& p, bool flip, float* mout) {
    const int nx = SEQ_L * D_MODEL;
    if (flip)
      cvt_h_fliprows_kernel<<<blocks4(nx), TB, 0, stream>>>(x, xh, SEQ_L, D_MODEL);
    else
      cvt_h_kernel<<<blocks4(nx), TB, 0, stream>>>(x, xh, nx / 4);
    cvt_h_kernel<<<blocks4(2 * D_INNER * D_MODEL), TB, 0, stream>>>(p.in_w, inwh, 2 * D_INNER * D_MODEL / 4);
    cvt_h_kernel<<<blocks4(D_MODEL * D_INNER), TB, 0, stream>>>(p.out_w, outwh, D_MODEL * D_INNER / 4);
    cvt_h_kernel<<<blocks4(XPROJ_N * D_INNER), TB, 0, stream>>>(p.xproj_w, xprojh, XPROJ_N * D_INNER / 4);
    cvt_h_kernel<<<blocks4(D_INNER * DT_RANK), TB, 0, stream>>>(p.dt_w, dtwh, D_INNER * DT_RANK / 4);

    // xz = x @ in_w^T   (1024 x 4096, K=1024)
    gemm64_wmma_abT<<<dim3((2 * D_INNER) / 64, SEQ_L / 64), 32, 0, stream>>>(
        xh, inwh, xz, nullptr, SEQ_L, 2 * D_INNER, D_MODEL);

    conv_silu_kernel<<<blocks(SEQ_L * D_INNER), TB, 0, stream>>>(xz, p.conv_w, p.conv_b, u, uh);

    // x_dbl = u @ xproj_w^T   (1024 x 96, K=2048) -- N=96, use 32x32 kernel
    gemm_wmma_abT<<<dim3(XPROJ_N / 32, SEQ_L / 32), 32, 0, stream>>>(
        uh, xprojh, xdbl, nullptr, SEQ_L, XPROJ_N, D_INNER);

    slice_dt_kernel<<<blocks(SEQ_L * DT_RANK), TB, 0, stream>>>(xdbl, dth);

    // dtlin = dt @ dt_w^T   (1024 x 2048, K=64); softplus(.+dt_b) fused in scan
    gemm64_wmma_abT<<<dim3(D_INNER / 64, SEQ_L / 64), 32, 0, stream>>>(
        dth, dtwh, dtlin, nullptr, SEQ_L, D_INNER, DT_RANK);

    scan_kernel<<<D_INNER / TB, TB, 0, stream>>>(dtlin, p.dt_b, p.Alog, u, xdbl, y);

    gate_kernel<<<blocks(SEQ_L * D_INNER), TB, 0, stream>>>(y, u, p.Dp, xz, yzh);

    // mout = yz @ out_w^T   (1024 x 1024, K=2048)
    gemm64_wmma_abT<<<dim3(D_MODEL / 64, SEQ_L / 64), 32, 0, stream>>>(
        yzh, outwh, mout, nullptr, SEQ_L, D_MODEL, D_INNER);
  };

  run_dir(fw, false, mout_fw);
  run_dir(bw, true,  mout_bw);

  concat_kernel<<<blocks(SEQ_L * D_MODEL), TB, 0, stream>>>(mout_fw, mout_bw, x, conch);
  cvt_h_kernel<<<blocks4(D_MODEL * 2 * D_MODEL), TB, 0, stream>>>(proj_w, projwh, D_MODEL * 2 * D_MODEL / 4);

  // pout = concat @ proj_w^T + proj_b   (1024 x 1024, K=2048)
  gemm64_wmma_abT<<<dim3(D_MODEL / 64, SEQ_L / 64), 32, 0, stream>>>(
      conch, projwh, pout, proj_b, SEQ_L, D_MODEL, 2 * D_MODEL);

  layernorm_kernel<<<SEQ_L, TB, 0, stream>>>(pout, ln_w, ln_b, (float*)d_out);
}
